// KnnXLMultiHeadsAttention_88416196756145
// MI455X (gfx1250) — matrix operand order repaired and checked by reading, not compiled, not run on hardware
//
#include <hip/hip_runtime.h>
#include <hip/hip_bf16.h>
#include <math.h>

#define B_    4
#define S_    512
#define E_    1024
#define H_    16
#define D_    64
#define HD_   1024
#define XL_   512
#define J_    (XL_ + S_)
#define TOPK_ 8
#define MEXT_ (S_ * TOPK_)   // 4096 external keys per query row

typedef __attribute__((ext_vector_type(16))) __bf16         v16bf;
typedef __attribute__((ext_vector_type(8)))  float          v8f;
typedef __attribute__((ext_vector_type(8)))  unsigned short u16x8;

union AFrag {
  v16bf v;
  u16x8 h[2];
  __bf16 e[16];
  unsigned short u[16];
};

__device__ inline __bf16 f2bf(float f) { return (__bf16)f; }

// A-matrix (16x32 bf16) lane layout: lanes 0-15 hold K={0..7,16..23},
// lanes 16-31 hold K={8..15,24..31}; two contiguous 8-element runs.
__device__ inline void load_a_bf16(AFrag& f, const __bf16* p, int lane) {
  int hi = (lane >> 4) & 1;
  f.h[0] = *(const u16x8*)(p + hi * 8);
  f.h[1] = *(const u16x8*)(p + 16 + hi * 8);
}
// B-matrix (32x16 bf16) lane layout: lanes 0-15 hold K={0..7,16..23},
// lanes 16-31 hold K={8..15,24..31}; two contiguous 8-element runs.
__device__ inline void load_b_bf16(AFrag& f, const __bf16* p, int lane) {
  const u16x8* q = (const u16x8*)(p + (((lane >> 4) & 1) << 3));
  f.h[0] = q[0];
  f.h[1] = q[2];
}

__device__ inline v8f wmma_bf16(const AFrag& a, const AFrag& b, v8f c) {
  return __builtin_amdgcn_wmma_f32_16x16x32_bf16(false, a.v, false, b.v,
                                                 (short)0, c, false, false);
}

// Async copy 16B / lane from global to LDS (tracked by ASYNCcnt).
// lds_off is the byte offset inside the workgroup LDS segment.
__device__ inline void async_ld16(unsigned lds_off, const void* g) {
  asm volatile("global_load_async_to_lds_b128 %0, %1, off"
               :: "v"(lds_off), "v"((unsigned long long)(size_t)g)
               : "memory");
}

// ---------------------------------------------------------------------------
// fp32 -> bf16 bulk convert, 8 elements / thread
// ---------------------------------------------------------------------------
__global__ void __launch_bounds__(256) cvt_bf16_kernel(
    const float* __restrict__ src, __bf16* __restrict__ dst, int n8) {
  int i = blockIdx.x * blockDim.x + threadIdx.x;
  if (i >= n8) return;
  const float4* s = (const float4*)src + 2 * (size_t)i;
  float4 a = s[0], b = s[1];
  union { u16x8 v; __bf16 e[8]; } o;
  o.e[0] = f2bf(a.x); o.e[1] = f2bf(a.y); o.e[2] = f2bf(a.z); o.e[3] = f2bf(a.w);
  o.e[4] = f2bf(b.x); o.e[5] = f2bf(b.y); o.e[6] = f2bf(b.z); o.e[7] = f2bf(b.w);
  ((u16x8*)dst)[i] = o.v;
}

// ---------------------------------------------------------------------------
// Pack kernels: build bf16 operand layouts for the attention kernels
// ---------------------------------------------------------------------------
__global__ void __launch_bounds__(256) pack_k_main_kernel(
    const float* __restrict__ xlmem, const float* __restrict__ k,
    __bf16* __restrict__ kbf) {
  int i = blockIdx.x * blockDim.x + threadIdx.x;   // B*H*J*D
  int d  = i & (D_ - 1);
  int jj = (i >> 6) & (J_ - 1);
  int h  = (i >> 16) & (H_ - 1);
  int b  = i >> 20;
  float s = (jj < XL_)
    ? xlmem[(((size_t)(b * XL_ + jj)) * 2) * HD_ + h * D_ + d]
    : k[((size_t)(b * S_ + (jj - XL_))) * HD_ + h * D_ + d];
  kbf[i] = f2bf(s);
}

__global__ void __launch_bounds__(256) pack_v_main_kernel(
    const float* __restrict__ xlmem, const float* __restrict__ v,
    __bf16* __restrict__ vbt) {
  int i = blockIdx.x * blockDim.x + threadIdx.x;   // B*H*D*J
  int jj = i & (J_ - 1);
  int d  = (i >> 10) & (D_ - 1);
  int h  = (i >> 16) & (H_ - 1);
  int b  = i >> 20;
  float s = (jj < XL_)
    ? xlmem[(((size_t)(b * XL_ + jj)) * 2 + 1) * HD_ + h * D_ + d]
    : v[((size_t)(b * S_ + (jj - XL_))) * HD_ + h * D_ + d];
  vbt[i] = f2bf(s);
}

__global__ void __launch_bounds__(256) pack_k_ext_kernel(
    const float* __restrict__ kvext, __bf16* __restrict__ keb) {
  int i = blockIdx.x * blockDim.x + threadIdx.x;   // B*H*M*D
  int d = i & (D_ - 1);
  int m = (i >> 6) & (MEXT_ - 1);
  int h = (i >> 18) & (H_ - 1);
  int b = i >> 22;
  keb[i] = f2bf(kvext[((((size_t)(b * S_ + (m >> 3))) * TOPK_ + (m & 7)) * 2) * HD_
                      + h * D_ + d]);
}

__global__ void __launch_bounds__(256) pack_v_ext_kernel(
    const float* __restrict__ kvext, __bf16* __restrict__ veb) {
  int i = blockIdx.x * blockDim.x + threadIdx.x;   // B*H*D*M
  int m = i & (MEXT_ - 1);
  int d = (i >> 12) & (D_ - 1);
  int h = (i >> 18) & (H_ - 1);
  int b = i >> 22;
  veb[i] = f2bf(kvext[((((size_t)(b * S_ + (m >> 3))) * TOPK_ + (m & 7)) * 2 + 1) * HD_
                      + h * D_ + d]);
}

// ---------------------------------------------------------------------------
// C[m,n] = sum_k A[m,k]*W[n,k] + bias[n]; one 16x64 C strip per wave
// ---------------------------------------------------------------------------
__global__ void __launch_bounds__(256) gemm_bias_kernel(
    const __bf16* __restrict__ A, const __bf16* __restrict__ W,
    const float* __restrict__ bias, float* __restrict__ C,
    int M, int N, int K) {
  int lane = threadIdx.x & 31;
  int col  = lane & 15;
  int gw   = blockIdx.x * (blockDim.x >> 5) + (threadIdx.x >> 5);
  int ngrp = N >> 6;
  int tm = gw / ngrp, tg = gw % ngrp;
  if (tm >= (M >> 4)) return;
  int m0 = tm << 4, n0 = tg << 6;
  const __bf16* arow  = A + (size_t)(m0 + col) * K;
  const __bf16* brow0 = W + (size_t)(n0 + col) * K;
  const __bf16* brow1 = brow0 + (size_t)16 * K;
  const __bf16* brow2 = brow0 + (size_t)32 * K;
  const __bf16* brow3 = brow0 + (size_t)48 * K;
  v8f acc0 = {0.f,0.f,0.f,0.f,0.f,0.f,0.f,0.f};
  v8f acc1 = acc0, acc2 = acc0, acc3 = acc0;
  for (int k0 = 0; k0 < K; k0 += 32) {
    AFrag a, b0, b1, b2, b3;
    load_a_bf16(a, arow + k0, lane);
    load_b_bf16(b0, brow0 + k0, lane);
    load_b_bf16(b1, brow1 + k0, lane);
    load_b_bf16(b2, brow2 + k0, lane);
    load_b_bf16(b3, brow3 + k0, lane);
    acc0 = wmma_bf16(a, b0, acc0);
    acc1 = wmma_bf16(a, b1, acc1);
    acc2 = wmma_bf16(a, b2, acc2);
    acc3 = wmma_bf16(a, b3, acc3);
  }
  int hi = (lane >> 4) & 1;
  float bb0 = bias[n0 + col];
  float bb1 = bias[n0 + 16 + col];
  float bb2 = bias[n0 + 32 + col];
  float bb3 = bias[n0 + 48 + col];
  for (int r = 0; r < 8; ++r) {
    int m = m0 + r + hi * 8;
    float* cp = C + (size_t)m * N + n0 + col;
    cp[0]  = acc0[r] + bb0;
    cp[16] = acc1[r] + bb1;
    cp[32] = acc2[r] + bb2;
    cp[48] = acc3[r] + bb3;
  }
}

// ---------------------------------------------------------------------------
// Row-wise L2 normalize (in place)
// ---------------------------------------------------------------------------
__global__ void __launch_bounds__(256) l2norm_kernel(float* __restrict__ x, int ncols) {
  __shared__ float red[256];
  float* p = x + (size_t)blockIdx.x * ncols;
  int tid = threadIdx.x;
  float ss = 0.f;
  for (int i = tid; i < ncols; i += 256) { float t = p[i]; ss += t * t; }
  red[tid] = ss;
  __syncthreads();
  for (int s = 128; s > 0; s >>= 1) {
    if (tid < s) red[tid] += red[tid + s];
    __syncthreads();
  }
  float inv = 1.f / fmaxf(sqrtf(red[0]), 1e-12f);
  for (int i = tid; i < ncols; i += 256) p[i] *= inv;
}

// ---------------------------------------------------------------------------
// Flash attention with async global->LDS staging (double buffered).
// A block = 4 waves sharing one (b,h); each wave owns one 16-row i-tile.
// LDS arena (offset 0 of the segment, single __shared__ symbol):
//   [0      ..  4096) buf0 k-tile  (32 j  x 64 d) bf16
//   [4096   ..  8192) buf0 vT-tile (64 d  x 32 j) bf16
//   [8192   .. 16384) buf1 (same pair)
//   [16384  .. 20480) per-wave P tiles (4 x 16x32 bf16)
// MODE 0: rel_pos + causal mask (Jd = J_);  MODE 1: external kNN (Jd = MEXT_)
// ---------------------------------------------------------------------------
#define SC2_ (0.125f * 1.44269504088896340736f)   // D^-0.5 * log2(e)

template <int MODE>
__global__ void __launch_bounds__(128) attn_kernel(
    const __bf16* __restrict__ qbf,   // (B,S,HD)
    const __bf16* __restrict__ kbf,   // (B,H,Jd,D)
    const __bf16* __restrict__ vbt,   // (B,H,D,Jd)
    const float*  __restrict__ relpos,// (1,H,S,J_)  (MODE 0 only)
    float* __restrict__ outbuf,       // (B,S,HD)
    int Jd) {
  __shared__ __attribute__((aligned(16))) unsigned char smem[20480];

  int tid  = threadIdx.x;
  int lane = tid & 31;
  int wv   = tid >> 5;
  int bi   = blockIdx.x;               // 512 blocks = B*H*8
  int g    = bi & 7;                   // group of 4 i-tiles
  int h    = (bi >> 3) & (H_ - 1);
  int b    = bi >> 7;
  int i0   = (g * 4 + wv) << 4;
  int col  = lane & 15;
  int hi   = (lane >> 4) & 1;

  const __bf16* kb = kbf + ((size_t)(b * H_ + h)) * Jd * D_;
  const __bf16* vb = vbt + ((size_t)(b * H_ + h)) * D_ * Jd;

  AFrag aq0, aq1;
  {
    const __bf16* qp = qbf + ((size_t)(b * S_ + i0 + col)) * HD_ + h * D_;
    load_a_bf16(aq0, qp, lane);
    load_a_bf16(aq1, qp + 32, lane);
  }

  v8f o0 = {0.f,0.f,0.f,0.f,0.f,0.f,0.f,0.f};
  v8f o1 = o0, o2 = o0, o3 = o0;
  float m[8], l[8];
  for (int r = 0; r < 8; ++r) { m[r] = -1e30f; l[r] = 0.f; }

  // uniform trip count for the whole block (largest i-tile; surplus tiles
  // are fully masked for the smaller i0's -> exp() == 0, a numeric no-op)
  int i0max  = (g * 4 + 3) << 4;
  int jtot   = (MODE == 0) ? (i0max + XL_ + 16) : Jd;
  int nsteps = (jtot + 31) >> 5;

  // per-thread staging assignments (16B per async issue)
  int krow = tid >> 3, kseg = tid & 7;   // k: 32 rows x 128B
  int vrow = tid >> 2, vseg = tid & 3;   // vT: 64 rows x 64B

  // prologue: stage step 0 into buffer 0
  {
    unsigned bb = 0;
    async_ld16(bb + krow * 128 + kseg * 16,           kb + (size_t)krow * D_ + kseg * 8);
    async_ld16(bb + (krow + 16) * 128 + kseg * 16,    kb + (size_t)(krow + 16) * D_ + kseg * 8);
    async_ld16(bb + 4096 + vrow * 64 + vseg * 16,     vb + (size_t)vrow * Jd + vseg * 8);
    async_ld16(bb + 4096 + (vrow + 32) * 64 + vseg * 16,
               vb + (size_t)(vrow + 32) * Jd + vseg * 8);
  }

  __bf16* pb = (__bf16*)(smem + 16384) + wv * 512;

  for (int js = 0; js < nsteps; ++js) {
    int j0 = js << 5;
    if (js + 1 < nsteps) {              // stage next step into other buffer
      int j1 = j0 + 32;
      unsigned bb = (js & 1) ? 0u : 8192u;
      async_ld16(bb + krow * 128 + kseg * 16,        kb + (size_t)(j1 + krow) * D_ + kseg * 8);
      async_ld16(bb + (krow + 16) * 128 + kseg * 16, kb + (size_t)(j1 + krow + 16) * D_ + kseg * 8);
      async_ld16(bb + 4096 + vrow * 64 + vseg * 16,  vb + (size_t)vrow * Jd + j1 + vseg * 8);
      async_ld16(bb + 4096 + (vrow + 32) * 64 + vseg * 16,
                 vb + (size_t)(vrow + 32) * Jd + j1 + vseg * 8);
      asm volatile("s_wait_asynccnt 4" ::: "memory");   // older group complete
    } else {
      asm volatile("s_wait_asynccnt 0" ::: "memory");
    }
    __syncthreads();   // whole tile (all waves' parts) visible

    const __bf16* kl = (const __bf16*)(smem + ((js & 1) ? 8192u : 0u));
    const __bf16* vl = kl + 2048;       // +4096 bytes

    v8f sc[2];
    for (int t = 0; t < 2; ++t) {
      const __bf16* kp = kl + (size_t)(t * 16 + col) * 64;
      AFrag bk0, bk1;
      load_b_bf16(bk0, kp, lane);
      load_b_bf16(bk1, kp + 32, lane);
      v8f s = {0.f,0.f,0.f,0.f,0.f,0.f,0.f,0.f};
      s = wmma_bf16(aq0, bk0, s);
      s = wmma_bf16(aq1, bk1, s);
      int jj = j0 + t * 16 + col;
      if (MODE == 0) {
        for (int r = 0; r < 8; ++r) {
          int irow = i0 + r + hi * 8;
          float sv = (s[r] + relpos[((size_t)(h * S_ + irow)) * J_ + jj]) * SC2_;
          if (jj > irow + XL_) sv = -1e30f;   // causal: triu(k = j-i+1)
          s[r] = sv;
        }
      } else {
        for (int r = 0; r < 8; ++r) s[r] *= SC2_;
      }
      sc[t] = s;
    }

    // online softmax in exp2 domain (rows striped across lane halves)
    float p0[8], p1[8];
    for (int r = 0; r < 8; ++r) {
      float rm = fmaxf(sc[0][r], sc[1][r]);
      for (int off = 1; off < 16; off <<= 1)
        rm = fmaxf(rm, __shfl_xor(rm, off, 32));
      float mn    = fmaxf(m[r], rm);
      float alpha = exp2f(m[r] - mn);
      m[r] = mn;
      float a0 = exp2f(sc[0][r] - mn);
      float a1 = exp2f(sc[1][r] - mn);
      float rs = a0 + a1;
      for (int off = 1; off < 16; off <<= 1)
        rs += __shfl_xor(rs, off, 32);
      l[r] = l[r] * alpha + rs;
      o0[r] *= alpha; o1[r] *= alpha; o2[r] *= alpha; o3[r] *= alpha;
      p0[r] = a0; p1[r] = a1;
    }

    // C-layout P -> A-layout through per-wave LDS tile (intra-wave in-order)
    for (int r = 0; r < 8; ++r) {
      int row = r + hi * 8;
      pb[row * 32 + col]      = f2bf(p0[r]);
      pb[row * 32 + 16 + col] = f2bf(p1[r]);
    }
    AFrag ap;
    ap.h[0] = *(const u16x8*)&pb[col * 32 + hi * 8];
    ap.h[1] = *(const u16x8*)&pb[col * 32 + 16 + hi * 8];

    // O += P(16x32) * V(32x64) from the staged vT tile
    {
      AFrag bv0, bv1, bv2, bv3;
      load_b_bf16(bv0, vl + (size_t)(col)      * 32, lane);
      load_b_bf16(bv1, vl + (size_t)(16 + col) * 32, lane);
      load_b_bf16(bv2, vl + (size_t)(32 + col) * 32, lane);
      load_b_bf16(bv3, vl + (size_t)(48 + col) * 32, lane);
      o0 = wmma_bf16(ap, bv0, o0);
      o1 = wmma_bf16(ap, bv1, o1);
      o2 = wmma_bf16(ap, bv2, o2);
      o3 = wmma_bf16(ap, bv3, o3);
    }
    __syncthreads();   // all waves done reading before buffer is re-staged
  }

  for (int r = 0; r < 8; ++r) {
    float inv  = 1.f / l[r];
    int   irow = i0 + r + hi * 8;
    float* op = outbuf + ((size_t)(b * S_ + irow)) * HD_ + h * D_ + col;
    op[0]  = o0[r] * inv;
    op[16] = o1[r] * inv;
    op[32] = o2[r] * inv;
    op[48] = o3[r] * inv;
  }
}

// fused = qkv * sigmoid(g) + qkv_ext * (1 - sigmoid(g)), output in bf16
__global__ void __launch_bounds__(256) fuse_gate_kernel(
    const float* __restrict__ a, const float* __restrict__ e,
    const float* __restrict__ gate, __bf16* __restrict__ out, int n) {
  int i = blockIdx.x * blockDim.x + threadIdx.x;
  if (i >= n) return;
  int h = (i & (HD_ - 1)) >> 6;
  float g = 1.f / (1.f + __expf(-gate[h]));
  out[i] = f2bf(a[i] * g + e[i] * (1.f - g));
}

// current_kv (B,S,2,HD) = stack(k_norm, v)
__global__ void __launch_bounds__(256) write_kv_kernel(
    const float* __restrict__ k, const float* __restrict__ v,
    float* __restrict__ out2, int n) {
  int i = blockIdx.x * blockDim.x + threadIdx.x;
  if (i >= n) return;
  int bs = i >> 10;
  int hd = i & (HD_ - 1);
  out2[((size_t)bs * 2) * HD_ + hd]     = k[i];
  out2[((size_t)bs * 2 + 1) * HD_ + hd] = v[i];
}

extern "C" void kernel_launch(void* const* d_in, const int* in_sizes, int n_in,
                              void* d_out, int out_size, void* d_ws, size_t ws_size,
                              hipStream_t stream) {
  const float* input  = (const float*)d_in[0];
  const float* relpos = (const float*)d_in[1];
  const float* xlmem  = (const float*)d_in[2];
  const float* kvext  = (const float*)d_in[3];
  const float* Wq = (const float*)d_in[4];
  const float* bq = (const float*)d_in[5];
  const float* Wk = (const float*)d_in[6];
  const float* bk = (const float*)d_in[7];
  const float* Wv = (const float*)d_in[8];
  const float* bvp = (const float*)d_in[9];
  const float* Wo = (const float*)d_in[10];
  const float* bo = (const float*)d_in[11];
  const float* gate = (const float*)d_in[12];
  float* out = (float*)d_out;

  const int    M   = B_ * S_;                 // 2048
  const size_t SZ  = (size_t)M * HD_;         // 2M elements
  const size_t WSZ = (size_t)HD_ * E_;        // 1M elements (weights)
  const size_t KVM = (size_t)B_ * H_ * J_ * D_;     // 4M  (main k/v)
  const size_t KVE = (size_t)B_ * H_ * MEXT_ * D_;  // 16M (ext  k/v)

  char* w = (char*)d_ws;
  float*  q    = (float*)w;            w += SZ  * 4;
  float*  k    = (float*)w;            w += SZ  * 4;
  float*  v    = (float*)w;            w += SZ  * 4;
  float*  qA   = (float*)w;            w += SZ  * 4;
  float*  qE   = (float*)w;            w += SZ  * 4;
  __bf16* inbf = (__bf16*)w;           w += SZ  * 2;
  __bf16* wqb  = (__bf16*)w;           w += WSZ * 2;
  __bf16* wkb  = (__bf16*)w;           w += WSZ * 2;
  __bf16* wvb  = (__bf16*)w;           w += WSZ * 2;
  __bf16* wob  = (__bf16*)w;           w += WSZ * 2;
  __bf16* qbf  = (__bf16*)w;           w += SZ  * 2;
  __bf16* kbf  = (__bf16*)w;           w += KVM * 2;
  __bf16* vbt  = (__bf16*)w;           w += KVM * 2;
  __bf16* keb  = (__bf16*)w;           w += KVE * 2;
  __bf16* veb  = (__bf16*)w;           w += KVE * 2;
  __bf16* fzb  = (__bf16*)w;           w += SZ  * 2;

  cvt_bf16_kernel<<<(int)(SZ / 8 / 256), 256, 0, stream>>>(input, inbf, (int)(SZ / 8));
  cvt_bf16_kernel<<<(int)(WSZ / 8 / 256), 256, 0, stream>>>(Wq, wqb, (int)(WSZ / 8));
  cvt_bf16_kernel<<<(int)(WSZ / 8 / 256), 256, 0, stream>>>(Wk, wkb, (int)(WSZ / 8));
  cvt_bf16_kernel<<<(int)(WSZ / 8 / 256), 256, 0, stream>>>(Wv, wvb, (int)(WSZ / 8));
  cvt_bf16_kernel<<<(int)(WSZ / 8 / 256), 256, 0, stream>>>(Wo, wob, (int)(WSZ / 8));

  const int gemm_blocks = (M / 16) * (HD_ / 64) / 8;   // 8 waves per block
  gemm_bias_kernel<<<gemm_blocks, 256, 0, stream>>>(inbf, wqb, bq, q, M, HD_, E_);
  gemm_bias_kernel<<<gemm_blocks, 256, 0, stream>>>(inbf, wkb, bk, k, M, HD_, E_);
  gemm_bias_kernel<<<gemm_blocks, 256, 0, stream>>>(inbf, wvb, bvp, v, M, HD_, E_);

  l2norm_kernel<<<M, 256, 0, stream>>>(q, HD_);
  l2norm_kernel<<<M, 256, 0, stream>>>(k, HD_);

  cvt_bf16_kernel<<<(int)(SZ / 8 / 256), 256, 0, stream>>>(q, qbf, (int)(SZ / 8));
  pack_k_main_kernel<<<(int)(KVM / 256), 256, 0, stream>>>(xlmem, k, kbf);
  pack_v_main_kernel<<<(int)(KVM / 256), 256, 0, stream>>>(xlmem, v, vbt);
  pack_k_ext_kernel<<<(int)(KVE / 256), 256, 0, stream>>>(kvext, keb);
  pack_v_ext_kernel<<<(int)(KVE / 256), 256, 0, stream>>>(kvext, veb);

  attn_kernel<0><<<512, 128, 0, stream>>>(qbf, kbf, vbt, relpos, qA, J_);
  attn_kernel<1><<<512, 128, 0, stream>>>(qbf, keb, veb, nullptr, qE, MEXT_);

  int n = (int)SZ;
  fuse_gate_kernel<<<(n + 255) / 256, 256, 0, stream>>>(qA, qE, gate, fzb, n);

  gemm_bias_kernel<<<gemm_blocks, 256, 0, stream>>>(fzb, wob, bo, out, M, E_, HD_);
  write_kv_kernel<<<(n + 255) / 256, 256, 0, stream>>>(k, v, out + SZ, n);
}
